// _ParticleNet_81157702025820
// MI455X (gfx1250) — compile-verified, matrix operand order checked by reading, not run on hardware
//
#include <hip/hip_runtime.h>
#include <hip/hip_bf16.h>

// ---------------------------------------------------------------------------
// ParticleNet (DGCNN) forward for MI455X / gfx1250.
// Edge-MLP GEMMs (~152 GFLOP) -> v_wmma_f32_16x16x32_f16, BN folded into f16
// weights, gather fused into layer-1 GEMM (compile-time DIN => branch-free),
// LDS double-buffering with CDNA5 async global->LDS staging (ASYNCcnt).
// Wave tile = 32x64 (2 A sub-tiles share 4 B-fragments): 8 WMMA per 12
// ds_load_b128 per K-step.
// ---------------------------------------------------------------------------

typedef __attribute__((ext_vector_type(16))) _Float16 v16h;
typedef __attribute__((ext_vector_type(8)))  _Float16 v8h;
typedef __attribute__((ext_vector_type(8)))  float    v8f;
typedef __attribute__((ext_vector_type(4)))  int      v4i;

#define NB   128          // batches
#define NP   128          // points per batch
#define KNN  16           // neighbors
#define EDGES (NB*NP*KNN) // 262144 edge rows

#ifndef __has_builtin
#define __has_builtin(x) 0
#endif
#if __has_builtin(__builtin_amdgcn_global_load_async_to_lds_b128)
#define USE_ASYNC_LDS 1
#else
#define USE_ASYNC_LDS 0
#endif

// 16-byte global -> LDS stage (async CDNA5 path; ASYNCcnt-tracked)
__device__ __forceinline__ void copy16(_Float16* l, const _Float16* g) {
#if USE_ASYNC_LDS
  __builtin_amdgcn_global_load_async_to_lds_b128((v4i*)g, (v4i*)l, 0, 0);
#else
  *(v8h*)l = *(const v8h*)g;
#endif
}

__device__ __forceinline__ void wait_stage() {
#if USE_ASYNC_LDS
#if __has_builtin(__builtin_amdgcn_s_wait_asynccnt)
  __builtin_amdgcn_s_wait_asynccnt(0);
#else
  asm volatile("s_wait_asynccnt 0x0" ::: "memory");
#endif
#endif
}

// ---------------------------------------------------------------------------
// Fold BN into weights: relu( e @ (W*scale) + (b*scale + beta - mean*scale) ).
// Weights stored TRANSPOSED: wt[nc][kpad] (f16, K zero-padded to mult of 32).
// ---------------------------------------------------------------------------
__global__ void fold_kernel(const float* __restrict__ w, const float* __restrict__ b,
                            const float* __restrict__ gamma, const float* __restrict__ beta,
                            const float* __restrict__ mean, const float* __restrict__ var,
                            _Float16* __restrict__ wt, float* __restrict__ bias,
                            int din2, int kpad, int nc) {
  int t = blockIdx.x * blockDim.x + threadIdx.x;
  if (t >= kpad * nc) return;
  int n = t % nc, kp = t / nc;
  float sc = gamma[n] * rsqrtf(var[n] + 1e-5f);
  wt[(size_t)n * kpad + kp] = (kp < din2) ? (_Float16)(w[(size_t)kp * nc + n] * sc)
                                          : (_Float16)0.f;
  if (kp == 0) bias[n] = b[n] * sc + beta[n] - mean[n] * sc;
}

// ---------------------------------------------------------------------------
// KNN: one WG per batch, 128 threads (one per point), features in LDS as f16,
// register insertion-sort top-16.
// ---------------------------------------------------------------------------
__global__ __launch_bounds__(NP)
void knn_kernel(const float* __restrict__ x, int F, int D, int* __restrict__ idx) {
  extern __shared__ _Float16 sx[];   // [NP][D]
  const int b = blockIdx.x, i = threadIdx.x;
  for (int d = 0; d < D; ++d)
    sx[i * D + d] = (_Float16)x[((size_t)b * NP + i) * F + d];
  __syncthreads();

  float bd[KNN]; int bi[KNN];
#pragma unroll
  for (int s = 0; s < KNN; ++s) { bd[s] = 3.0e38f; bi[s] = 0; }

  for (int j = 0; j < NP; ++j) {
    float d2 = (j == i) ? 1.0e9f : 0.f;
    for (int d = 0; d < D; ++d) {
      float t = (float)sx[i * D + d] - (float)sx[j * D + d];
      d2 += t * t;
    }
    if (d2 < bd[KNN - 1]) {
      bd[KNN - 1] = d2; bi[KNN - 1] = j;
#pragma unroll
      for (int s = KNN - 1; s > 0; --s) {
        if (bd[s] < bd[s - 1]) {
          float td = bd[s]; bd[s] = bd[s - 1]; bd[s - 1] = td;
          int   ti = bi[s]; bi[s] = bi[s - 1]; bi[s - 1] = ti;
        }
      }
    }
  }
#pragma unroll
  for (int s = 0; s < KNN; ++s) idx[((size_t)b * NP + i) * KNN + s] = bi[s];
}

// ---------------------------------------------------------------------------
// WMMA fragment from one 32-half LDS row.
// 16-bit layout: lane<16 -> K {0..7,16..23}, lane>=16 -> K {8..15,24..31};
// each half-run contiguous -> two ds_load_b128.
// ---------------------------------------------------------------------------
__device__ __forceinline__ v16h make_frag(const _Float16* row32, int hi) {
  const v8h* p = (const v8h*)row32;
  v8h a = p[hi];
  v8h c = p[hi + 2];
  return __builtin_shufflevector(a, c, 0,1,2,3,4,5,6,7,8,9,10,11,12,13,14,15);
}

__device__ __forceinline__ v8f wmma_f16(v16h a, v16h b, v8f c) {
  return __builtin_amdgcn_wmma_f32_16x16x32_f16(false, a, false, b, (short)0, c,
                                                false, false);
}

// One K-step: wave computes 32 rows x 64 cols. Two A-frags share 4 B-frags.
__device__ __forceinline__ void mma_step2(const _Float16* sAw, const _Float16* sB,
                                          int lane, v8f acc[2][4]) {
  const int hi = lane >> 4;
  v16h af0 = make_frag(sAw + (lane & 15) * 32, hi);
  v16h af1 = make_frag(sAw + ((lane & 15) + 16) * 32, hi);
#pragma unroll
  for (int ct = 0; ct < 4; ++ct) {
    v16h bf = make_frag(sB + (ct * 16 + (lane & 15)) * 32, hi);
    acc[0][ct] = wmma_f16(af0, bf, acc[0][ct]);
    acc[1][ct] = wmma_f16(af1, bf, acc[1][ct]);
  }
}

template <int NC>
__device__ __forceinline__ void epilogue2(const v8f acc[2][4], _Float16* __restrict__ out,
                                          const float* __restrict__ bias,
                                          int rowBase, int colBase, int lane, int wave) {
  const int rsel = (lane >> 4) << 3;
#pragma unroll
  for (int m = 0; m < 2; ++m) {
#pragma unroll
    for (int ct = 0; ct < 4; ++ct) {
      int col = colBase + ct * 16 + (lane & 15);
      float bsc = bias[col];
      int rr = rowBase + wave * 32 + m * 16 + rsel;
#pragma unroll
      for (int v = 0; v < 8; ++v) {
        float f = acc[m][ct][v] + bsc;
        f = f > 0.f ? f : 0.f;
        out[(size_t)(rr + v) * NC + col] = (_Float16)f;
      }
    }
  }
}

// ---------------------------------------------------------------------------
// Layer-1 GEMM, gather fused, DIN compile-time => branch-free gather.
// Every thread: one full A row (32 cols) -> 4x ds_store_b128, plus 2 async
// B chunks. Macro-tile 128x64, wave tile 32x64.
// ---------------------------------------------------------------------------
template <int DIN, int KPAD, int NC>
__global__ __launch_bounds__(128)
void gemm_edge1_t(const float* __restrict__ x, const int* __restrict__ idx,
                  const _Float16* __restrict__ wt, const float* __restrict__ bias,
                  _Float16* __restrict__ out) {
  __shared__ _Float16 sA[128 * 32];
  __shared__ _Float16 sB[64 * 32];   // [col][K]
  const int t = threadIdx.x, lane = t & 31, wave = t >> 5;
  const int rowBase = blockIdx.x * 128, colBase = blockIdx.y * 64;

  const int gRow = rowBase + t;
  const int bb = gRow >> 11, rem = gRow & 2047, ii = rem >> 4;
  const int jj = idx[gRow];
  const float* xi = x + ((size_t)bb * NP + ii) * DIN;
  const float* xj = x + ((size_t)bb * NP + jj) * DIN;

  const int bc0 = t >> 2, bseg = t & 3;
  const _Float16* gb0 = wt + (size_t)(colBase + bc0) * KPAD + bseg * 8;
  const _Float16* gb1 = wt + (size_t)(colBase + bc0 + 32) * KPAD + bseg * 8;

  v8f acc[2][4] = {};
#pragma unroll
  for (int k0 = 0; k0 < KPAD; k0 += 32) {
    // B tile: 2 async 16B chunks per thread
    copy16(&sB[bc0 * 32 + bseg * 8], gb0 + k0);
    copy16(&sB[(bc0 + 32) * 32 + bseg * 8], gb1 + k0);
    // A tile: branch-free gather of one full 32-col row
    float v[32];
#pragma unroll
    for (int e = 0; e < 32; ++e) {
      const int c = k0 + e;           // compile-time per unrolled iteration
      if (c < DIN)           v[e] = xi[c];
      else if (c < 2 * DIN)  v[e] = xj[c - DIN] - xi[c - DIN];
      else                   v[e] = 0.f;
    }
#pragma unroll
    for (int q = 0; q < 4; ++q) {
      v8h h;
#pragma unroll
      for (int e = 0; e < 8; ++e) h[e] = (_Float16)v[q * 8 + e];
      *(v8h*)&sA[t * 32 + q * 8] = h;
    }
    wait_stage();
    __syncthreads();
    mma_step2(&sA[wave * 32 * 32], sB, lane, acc);
    __syncthreads();
  }
  epilogue2<NC>(acc, out, bias, rowBase, colBase, lane, wave);
}

// ---------------------------------------------------------------------------
// Layers 2/3: f16 GEMM + bias + relu, LDS double-buffered, K fully unrolled,
// next tile staged asynchronously while WMMAs consume current tile.
// Macro-tile 128x64, wave tile 32x64.
// ---------------------------------------------------------------------------
template <int KD, int NC>
__global__ __launch_bounds__(128)
void gemm_relu_t(const _Float16* __restrict__ A,
                 const _Float16* __restrict__ wt,
                 const float* __restrict__ bias,
                 _Float16* __restrict__ out) {
  __shared__ _Float16 sA[2][128 * 32];
  __shared__ _Float16 sB[2][64 * 32];
  constexpr int S = KD / 32;
  const int t = threadIdx.x, lane = t & 31, wave = t >> 5;
  const int rowBase = blockIdx.x * 128, colBase = blockIdx.y * 64;
  const int r0 = t >> 2, seg = t & 3;

  const _Float16* ga[4];
  int la[4];
#pragma unroll
  for (int i = 0; i < 4; ++i) {
    ga[i] = A + (size_t)(rowBase + r0 + 32 * i) * KD + seg * 8;
    la[i] = (r0 + 32 * i) * 32 + seg * 8;
  }
  const _Float16* gb0 = wt + (size_t)(colBase + r0) * KD + seg * 8;
  const _Float16* gb1 = wt + (size_t)(colBase + r0 + 32) * KD + seg * 8;
  const int lb0 = r0 * 32 + seg * 8, lb1 = (r0 + 32) * 32 + seg * 8;

#pragma unroll
  for (int i = 0; i < 4; ++i) copy16(&sA[0][la[i]], ga[i]);
  copy16(&sB[0][lb0], gb0); copy16(&sB[0][lb1], gb1);

  v8f acc[2][4] = {};
#pragma unroll
  for (int s = 0; s < S; ++s) {
    const int p = s & 1;
    wait_stage();
    __syncthreads();                       // tile s visible to all waves
    if (s + 1 < S) {                       // stage next tile during compute
      const int k = (s + 1) * 32;
#pragma unroll
      for (int i = 0; i < 4; ++i) copy16(&sA[1 - p][la[i]], ga[i] + k);
      copy16(&sB[1 - p][lb0], gb0 + k); copy16(&sB[1 - p][lb1], gb1 + k);
    }
    mma_step2(&sA[p][wave * 32 * 32], sB[p], lane, acc);
    __syncthreads();
  }
  epilogue2<NC>(acc, out, bias, rowBase, colBase, lane, wave);
}

// ---------------------------------------------------------------------------
// Mean over K neighbors + concat [conv_out | x_old] -> new f32 features.
// ---------------------------------------------------------------------------
__global__ void reduce_concat(const _Float16* __restrict__ h, int L,
                              const float* __restrict__ xold, int Fold,
                              float* __restrict__ xnew) {
  const int Fnew = L + Fold;
  const int tot = NB * NP * Fnew;
  int t = blockIdx.x * blockDim.x + threadIdx.x;
  if (t >= tot) return;
  int c = t % Fnew;
  int p = t / Fnew;                       // b*NP + n
  if (c < L) {
    float s = 0.f;
#pragma unroll
    for (int k = 0; k < KNN; ++k)
      s += (float)h[(size_t)(p * KNN + k) * L + c];
    xnew[t] = s * (1.f / KNN);
  } else {
    xnew[t] = xold[(size_t)p * Fold + (c - L)];
  }
}

// ---------------------------------------------------------------------------
// Global mean pool over N + FC1(451->256,relu) + FC2(256->5). One WG per batch.
// ---------------------------------------------------------------------------
__global__ __launch_bounds__(256)
void final_fc(const float* __restrict__ x451,
              const float* __restrict__ w1, const float* __restrict__ b1,
              const float* __restrict__ w2, const float* __restrict__ b2,
              float* __restrict__ out) {
  __shared__ float pooled[451];
  __shared__ float h1[256];
  const int b = blockIdx.x, t = threadIdx.x;
  for (int c = t; c < 451; c += 256) {
    float s = 0.f;
    for (int n = 0; n < NP; ++n) s += x451[((size_t)b * NP + n) * 451 + c];
    pooled[c] = s * (1.f / NP);
  }
  __syncthreads();
  float a = b1[t];
  for (int c = 0; c < 451; ++c) a += pooled[c] * w1[(size_t)c * 256 + t];
  h1[t] = a > 0.f ? a : 0.f;
  __syncthreads();
  if (t < 5) {
    float a2 = b2[t];
    for (int c = 0; c < 256; ++c) a2 += h1[c] * w2[c * 5 + t];
    out[b * 5 + t] = a2;
  }
}

// ---------------------------------------------------------------------------
extern "C" void kernel_launch(void* const* d_in, const int* in_sizes, int n_in,
                              void* d_out, int out_size, void* d_ws, size_t ws_size,
                              hipStream_t stream) {
  (void)out_size; (void)ws_size;
  // --- resolve inputs (insertion-order OR jax sorted-key flatten) ---
  const float *X, *W[9], *Bv[9], *Ga[9], *Be[9], *Me[9], *Va[9];
  const float *fc1w, *fc1b, *fc2w, *fc2b;
  const bool orderA = (n_in > 0 && in_sizes[0] == NB * NP * 3);
  if (orderA) {
    X = (const float*)d_in[0];
    for (int l = 0; l < 9; ++l) {
      int g = 1 + l * 6;
      W[l]  = (const float*)d_in[g];     Bv[l] = (const float*)d_in[g + 1];
      Ga[l] = (const float*)d_in[g + 2]; Be[l] = (const float*)d_in[g + 3];
      Me[l] = (const float*)d_in[g + 4]; Va[l] = (const float*)d_in[g + 5];
    }
    fc1w = (const float*)d_in[55]; fc1b = (const float*)d_in[56];
    fc2w = (const float*)d_in[57]; fc2b = (const float*)d_in[58];
  } else {
    for (int l = 0; l < 9; ++l) {
      int g = l * 6;
      Bv[l] = (const float*)d_in[g];     Be[l] = (const float*)d_in[g + 1];
      Ga[l] = (const float*)d_in[g + 2]; Me[l] = (const float*)d_in[g + 3];
      Va[l] = (const float*)d_in[g + 4]; W[l]  = (const float*)d_in[g + 5];
    }
    fc1b = (const float*)d_in[54]; fc1w = (const float*)d_in[55];
    fc2b = (const float*)d_in[56]; fc2w = (const float*)d_in[57];
    X = (const float*)d_in[58];
  }

  // --- workspace carve (~317 MB) ---
  char* p = (char*)d_ws;
  auto carve = [&](size_t bytes) -> char* {
    char* r = p; p += (bytes + 255) & ~(size_t)255; return r;
  };
  float* x67  = (float*)carve((size_t)NB * NP * 67  * 4);
  float* x195 = (float*)carve((size_t)NB * NP * 195 * 4);
  float* x451 = (float*)carve((size_t)NB * NP * 451 * 4);
  int*   idx  = (int*)  carve((size_t)EDGES * 4);
  static const int KPAD[9] = {32, 64, 64, 160, 128, 128, 416, 256, 256};
  static const int NCOL[9] = {64, 64, 64, 128, 128, 128, 256, 256, 256};
  static const int DIN2[9] = {6, 64, 64, 134, 128, 128, 390, 256, 256};
  _Float16* wt[9]; float* bs[9];
  for (int l = 0; l < 9; ++l) {
    wt[l] = (_Float16*)carve((size_t)KPAD[l] * NCOL[l] * 2);
    bs[l] = (float*)   carve((size_t)NCOL[l] * 4);
  }
  _Float16* hA = (_Float16*)carve((size_t)EDGES * 256 * 2);
  _Float16* hB = (_Float16*)carve((size_t)EDGES * 256 * 2);

  // --- fold BN into f16 transposed weights ---
  for (int l = 0; l < 9; ++l) {
    int tot = KPAD[l] * NCOL[l];
    fold_kernel<<<(tot + 255) / 256, 256, 0, stream>>>(
        W[l], Bv[l], Ga[l], Be[l], Me[l], Va[l], wt[l], bs[l],
        DIN2[l], KPAD[l], NCOL[l]);
  }

  // --- 3 EdgeConv blocks ---
  const float* xs[3] = {X, x67, x195};
  float*       xd[3] = {x67, x195, x451};
  const int Fv[3] = {3, 67, 195};
  const int Dv[3] = {2, 67, 195};
  for (int blk = 0; blk < 3; ++blk) {
    const int F = Fv[blk], L = NCOL[blk * 3];
    knn_kernel<<<NB, NP, (size_t)Dv[blk] * NP * sizeof(_Float16), stream>>>(
        xs[blk], F, Dv[blk], idx);
    dim3 g(EDGES / 128, L / 64);
    switch (blk) {
      case 0:
        gemm_edge1_t<3, 32, 64><<<g, 128, 0, stream>>>(xs[0], idx, wt[0], bs[0], hA);
        gemm_relu_t<64, 64><<<g, 128, 0, stream>>>(hA, wt[1], bs[1], hB);
        gemm_relu_t<64, 64><<<g, 128, 0, stream>>>(hB, wt[2], bs[2], hA);
        break;
      case 1:
        gemm_edge1_t<67, 160, 128><<<g, 128, 0, stream>>>(xs[1], idx, wt[3], bs[3], hA);
        gemm_relu_t<128, 128><<<g, 128, 0, stream>>>(hA, wt[4], bs[4], hB);
        gemm_relu_t<128, 128><<<g, 128, 0, stream>>>(hB, wt[5], bs[5], hA);
        break;
      default:
        gemm_edge1_t<195, 416, 256><<<g, 128, 0, stream>>>(xs[2], idx, wt[6], bs[6], hA);
        gemm_relu_t<256, 256><<<g, 128, 0, stream>>>(hA, wt[7], bs[7], hB);
        gemm_relu_t<256, 256><<<g, 128, 0, stream>>>(hB, wt[8], bs[8], hA);
        break;
    }
    const int tot = NB * NP * (L + F);
    reduce_concat<<<(tot + 255) / 256, 256, 0, stream>>>(hA, L, xs[blk], F, xd[blk]);
  }

  // --- pool + FCs ---
  final_fc<<<NB, 256, 0, stream>>>(x451, fc1w, fc1b, fc2w, fc2b, (float*)d_out);
}
